// Experiment1_670014898908
// MI455X (gfx1250) — matrix operand order repaired and checked by reading, not compiled, not run on hardware
//
#include <hip/hip_runtime.h>
#include <hip/hip_bf16.h>

typedef unsigned short u16;
typedef __attribute__((ext_vector_type(16))) __bf16 v16bf;
typedef __attribute__((ext_vector_type(8)))  float  v8f;
typedef __attribute__((ext_vector_type(4)))  unsigned int v4u;
typedef __attribute__((ext_vector_type(8)))  int   v8i_;
typedef __attribute__((ext_vector_type(4)))  int   v4i_;

#define R_TOK   3072          // B * TOKENS
#define BATCH   512
#define SEQL    720
#define KPAD    768           // 720 padded to multiple of 128 (TDM chunk)
#define DM      1024
#define DS      64
#define DTR     64
#define PRED    96

union Frag16 { v16bf v; uint4 q[2]; };

static __device__ __forceinline__ u16 f32_to_bf16(float f) {
    unsigned int u = __float_as_uint(f);
    u += 0x7FFFu + ((u >> 16) & 1u);        // round-to-nearest-even
    return (u16)(u >> 16);
}

// ---------------------------------------------------------------------------
// Tensor Data Mover: load a 2D tile [tile0 x 64 rows] of 2-byte elements into
// LDS. D# per CDNA5 ISA ch.8: group0 = {count, lds_addr, global_addr, type=2},
// group1 = {data_size=1(2B), tensor_dim0/1, tile_dim0/1, dim0_stride}.
// Rows >= t1 are beyond tensor_dim1 -> TDM zero-fills (handles ragged N edge).
// ---------------------------------------------------------------------------
static __device__ __forceinline__
void tdm_load_tile(unsigned lds_off, const u16* gsrc, int t0, int t1,
                   int tile0, int strideElems)
{
    unsigned long long ga = (unsigned long long)(const void*)gsrc;
    v4u g0;
    g0[0] = 1u;                                            // count=1, user mode
    g0[1] = lds_off;                                       // lds_addr (bytes)
    g0[2] = (unsigned)ga;                                  // global_addr[31:0]
    g0[3] = (unsigned)((ga >> 32) & 0x01FFFFFFu) | (2u << 30); // addr[56:32]|type=2
    unsigned T0 = (unsigned)t0, T1 = (unsigned)t1;
    unsigned long long S0 = (unsigned long long)(unsigned)strideElems;
    v8i_ g1;
    g1[0] = (int)(1u << 16);                               // data_size = 2 bytes
    g1[1] = (int)((T0 & 0xFFFFu) << 16);                   // tensor_dim0 lo
    g1[2] = (int)((T0 >> 16) | ((T1 & 0xFFFFu) << 16));    // dim0 hi | dim1 lo
    g1[3] = (int)((T1 >> 16) | ((unsigned)tile0 << 16));   // dim1 hi | tile_dim0
    g1[4] = (int)64u;                                      // tile_dim1=64, tile_dim2=0
    g1[5] = (int)(unsigned)(S0 & 0xFFFFFFFFu);             // dim0_stride lo
    g1[6] = (int)(unsigned)((S0 >> 32) & 0xFFFFu);         // dim0_stride hi
    g1[7] = 0;
    v4i_ gz = {0, 0, 0, 0};
#if defined(__clang_major__) && __clang_major__ >= 23
    v8i_ gz8 = {};
    __builtin_amdgcn_tensor_load_to_lds(g0, g1, gz, gz, gz8, 0);
#else
    __builtin_amdgcn_tensor_load_to_lds(g0, g1, gz, gz, 0);
#endif
}

// ---------------------------------------------------------------------------
// WMMA GEMM: C[M,N] = A[M,K](bf16,row-major) * W[N,K]^T (+bias)(+relu)(+accum)
// Block: 256 thr = 8 waves -> 128x64 tile; wave -> 32x32 (2x2 wmma frags).
// Weight tile staged into LDS by the TDM (double-buffered, s_wait_tensorcnt),
// shared by all 8 waves; A fragments stream from global.
// ---------------------------------------------------------------------------
template<bool BIAS, bool RELU, bool ACCUM, bool OUTF32, bool OUTBF16>
__global__ __launch_bounds__(256)
void wmma_gemm_kernel(const u16* __restrict__ A, const u16* __restrict__ W,
                      const float* __restrict__ bias,
                      float* __restrict__ Cf, u16* __restrict__ Cb,
                      int M, int N, int K, int kchunk)
{
    __shared__ u16 sB[2][64 * 128];     // 2 x 16KB double buffer

    const int lane = threadIdx.x & 31;
    const int wave = threadIdx.x >> 5;
    const int wm = wave >> 1, wn = wave & 1;
    const int bM = blockIdx.y * 128 + wm * 32;
    const int colBase = blockIdx.x * 64;
    const int bN = colBase + wn * 32;

    const int r  = lane & 15;
    const int hi = lane >> 4;

    // A fragment base pointers (rows bM+r, bM+16+r); ISA wave32 A layout:
    // halves 0..7 = K[kb..kb+7], halves 8..15 = K[kb+16..kb+23], kb = hi*8
    const u16* pa0 = A + (size_t)(bM + r)      * K + hi * 8;
    const u16* pa1 = A + (size_t)(bM + 16 + r) * K + hi * 8;

    const int nchunks = K / kchunk;
    const int remCols = N - colBase;

    if (wave == 0)
        tdm_load_tile((unsigned)(size_t)&sB[0][0], W + (size_t)colBase * K,
                      K, remCols, kchunk, K);

    v8f c00 = {}, c01 = {}, c10 = {}, c11 = {};
    for (int c = 0; c < nchunks; c++) {
        if (wave == 0) {
            if (c + 1 < nchunks)
                tdm_load_tile((unsigned)(size_t)&sB[(c + 1) & 1][0],
                              W + (size_t)colBase * K + (c + 1) * kchunk,
                              K - (c + 1) * kchunk, remCols, kchunk, K);
            if (c + 1 < nchunks) __builtin_amdgcn_s_wait_tensorcnt(1);
            else                 __builtin_amdgcn_s_wait_tensorcnt(0);
        }
        __syncthreads();
        // B fragment: col = bN+r(+16); halves 0..7 = K[kb2..kb2+7], halves 8..15 = K[kb2+16..kb2+23], kb2=hi*8
        const u16* lb  = &sB[c & 1][0];
        const u16* lb0 = lb + (size_t)(wn * 32 + r)      * kchunk + hi * 8;
        const u16* lb1 = lb + (size_t)(wn * 32 + 16 + r) * kchunk + hi * 8;
        const int kg = c * kchunk;
        for (int kk = 0; kk < kchunk; kk += 32) {
            Frag16 a0, a1, b0, b1;
            a0.q[0] = *(const uint4*)(pa0 + kg + kk);
            a0.q[1] = *(const uint4*)(pa0 + kg + kk + 16);
            a1.q[0] = *(const uint4*)(pa1 + kg + kk);
            a1.q[1] = *(const uint4*)(pa1 + kg + kk + 16);
            b0.q[0] = *(const uint4*)(lb0 + kk);
            b0.q[1] = *(const uint4*)(lb0 + kk + 16);
            b1.q[0] = *(const uint4*)(lb1 + kk);
            b1.q[1] = *(const uint4*)(lb1 + kk + 16);
            c00 = __builtin_amdgcn_wmma_f32_16x16x32_bf16(false, a0.v, false, b0.v, (short)0, c00, false, false);
            c01 = __builtin_amdgcn_wmma_f32_16x16x32_bf16(false, a0.v, false, b1.v, (short)0, c01, false, false);
            c10 = __builtin_amdgcn_wmma_f32_16x16x32_bf16(false, a1.v, false, b0.v, (short)0, c10, false, false);
            c11 = __builtin_amdgcn_wmma_f32_16x16x32_bf16(false, a1.v, false, b1.v, (short)0, c11, false, false);
        }
        __syncthreads();
    }

    auto store_frag = [&](v8f& c, int rowb, int colb) {
        int col  = colb + (lane & 15);
        int row0 = rowb + (lane >> 4) * 8;
        bool okc = (col < N);
        float bv = (BIAS && okc) ? bias[col] : 0.0f;
        #pragma unroll
        for (int i = 0; i < 8; i++) {
            int row = row0 + i;
            float v = c[i] + bv;
            if (okc) {
                size_t idx = (size_t)row * N + col;
                if (ACCUM) v += Cf[idx];
                if (RELU)  v = fmaxf(v, 0.0f);
                if (OUTF32)  Cf[idx] = v;
                if (OUTBF16) Cb[idx] = f32_to_bf16(v);
            }
        }
    };
    store_frag(c00, bM + 0,  bN + 0);
    store_frag(c01, bM + 0,  bN + 16);
    store_frag(c10, bM + 16, bN + 0);
    store_frag(c11, bM + 16, bN + 16);
}

// ---------------------------------------------------------------------------
// Weight / activation conversion helpers
// ---------------------------------------------------------------------------
__global__ void to_bf16_kernel(const float* __restrict__ s, u16* __restrict__ d, int n) {
    int i = blockIdx.x * 256 + threadIdx.x;
    if (i < n) d[i] = f32_to_bf16(s[i]);
}

__global__ void embw_pad_kernel(const float* __restrict__ s, u16* __restrict__ d) {
    int i = blockIdx.x * 256 + threadIdx.x;
    if (i >= DM * KPAD) return;
    int r = i / KPAD, c = i % KPAD;
    d[i] = (c < SEQL) ? f32_to_bf16(s[r * SEQL + c]) : (u16)0;
}

__global__ void dt_slice_kernel(const float* __restrict__ dbc, u16* __restrict__ dtb) {
    int i = blockIdx.x * 256 + threadIdx.x;
    if (i >= R_TOK * DTR) return;
    int r = i >> 6, c = i & 63;
    dtb[i] = f32_to_bf16(dbc[(size_t)r * 192 + c]);
}

// ---------------------------------------------------------------------------
// RevIN + inverted-token build: tok[b*6+n, 0:KPAD] bf16, stats[b][m0,m1,s0,s1]
// ---------------------------------------------------------------------------
__global__ __launch_bounds__(256)
void revin_tokens_kernel(const float* __restrict__ x_enc, const float* __restrict__ x_mark,
                         u16* __restrict__ tok, float* __restrict__ stats)
{
    __shared__ float red[4][256];
    __shared__ float ms[4];
    int b = blockIdx.x;
    const float* xe = x_enc + (size_t)b * SEQL * 2;
    float s0 = 0, s1 = 0, q0 = 0, q1 = 0;
    for (int i = threadIdx.x; i < SEQL; i += 256) {
        float a = xe[i * 2], c = xe[i * 2 + 1];
        s0 += a; q0 += a * a; s1 += c; q1 += c * c;
    }
    red[0][threadIdx.x] = s0; red[1][threadIdx.x] = s1;
    red[2][threadIdx.x] = q0; red[3][threadIdx.x] = q1;
    __syncthreads();
    for (int o = 128; o > 0; o >>= 1) {
        if ((int)threadIdx.x < o)
            for (int j = 0; j < 4; j++) red[j][threadIdx.x] += red[j][threadIdx.x + o];
        __syncthreads();
    }
    if (threadIdx.x == 0) {
        float m0 = red[0][0] / (float)SEQL, m1 = red[1][0] / (float)SEQL;
        float v0 = red[2][0] / (float)SEQL - m0 * m0;
        float v1 = red[3][0] / (float)SEQL - m1 * m1;
        ms[0] = m0; ms[1] = m1;
        ms[2] = sqrtf(v0 + 1e-5f); ms[3] = sqrtf(v1 + 1e-5f);
        stats[b * 4 + 0] = ms[0]; stats[b * 4 + 1] = ms[1];
        stats[b * 4 + 2] = ms[2]; stats[b * 4 + 3] = ms[3];
    }
    __syncthreads();
    const float* xm = x_mark + (size_t)b * SEQL * 4;
    for (int i = threadIdx.x; i < 6 * KPAD; i += 256) {
        int n = i / KPAD, c = i % KPAD;
        float v = 0.0f;
        if (c < SEQL) v = (n < 2) ? (xe[c * 2 + n] - ms[n]) / ms[2 + n]
                                  : xm[c * 4 + (n - 2)];
        tok[(size_t)(b * 6 + n) * KPAD + c] = f32_to_bf16(v);
    }
}

// ---------------------------------------------------------------------------
// Causal depthwise conv (kernel=2, over token axis T=6) + SiLU.
// dir=0: neighbor t-1 ; dir=1 (reversed-scan mamba): neighbor t+1.
// ---------------------------------------------------------------------------
__global__ __launch_bounds__(256)
void conv_silu_kernel(const float* __restrict__ xz, const float* __restrict__ cw,
                      const float* __restrict__ cb, float* __restrict__ xcf,
                      u16* __restrict__ xcb, int dir)
{
    int idx = blockIdx.x * 256 + threadIdx.x;
    int d = idx & (DM - 1);
    int row = idx >> 10;
    int b = row / 6, t = row % 6;
    int tn = dir ? t + 1 : t - 1;
    float xprev = 0.0f;
    if (tn >= 0 && tn < 6) xprev = xz[(size_t)(b * 6 + tn) * 2048 + d];
    float xcur = xz[(size_t)row * 2048 + d];
    float v = cb[d] + cw[d * 2 + 0] * xprev + cw[d * 2 + 1] * xcur;
    v = v / (1.0f + __expf(-v));                 // SiLU
    xcf[(size_t)row * DM + d] = v;
    xcb[(size_t)row * DM + d] = f32_to_bf16(v);
}

// ---------------------------------------------------------------------------
// Selective scan: one thread per (b,d). 64-wide state in registers, B/C in LDS.
// ---------------------------------------------------------------------------
__global__ __launch_bounds__(256)
void scan_kernel(const float* __restrict__ dbc, const float* __restrict__ dt_lin,
                 const float* __restrict__ xcf, const float* __restrict__ xz,
                 const float* __restrict__ A_log, const float* __restrict__ D_param,
                 u16* __restrict__ yb, int dir)
{
    __shared__ float sB[6][64];
    __shared__ float sC[6][64];
    int b = blockIdx.y;
    int d = blockIdx.x * 256 + threadIdx.x;
    for (int i = threadIdx.x; i < 6 * 128; i += 256) {
        int t = i >> 7, c = i & 127;
        float v = dbc[(size_t)(b * 6 + t) * 192 + 64 + c];
        if (c < 64) sB[t][c] = v; else sC[t][c - 64] = v;
    }
    __syncthreads();

    float negA[64];
    #pragma unroll
    for (int s = 0; s < 64; s++) negA[s] = -__expf(A_log[(size_t)d * 64 + s]);
    float st[64];
    #pragma unroll
    for (int s = 0; s < 64; s++) st[s] = 0.0f;
    float Dp = D_param[d];

    for (int tt = 0; tt < 6; tt++) {
        int t = dir ? (5 - tt) : tt;
        size_t row = (size_t)b * 6 + t;
        float dl  = dt_lin[row * DM + d];
        float dtv = (dl > 20.0f) ? dl : log1pf(__expf(dl));     // softplus
        float xcv = xcf[row * DM + d];
        float du  = dtv * xcv;
        float acc = 0.0f;
        #pragma unroll
        for (int s = 0; s < 64; s++) {
            float da = __expf(dtv * negA[s]);
            float ns = da * st[s] + du * sB[t][s];
            st[s] = ns;
            acc += ns * sC[t][s];
        }
        float y = acc + Dp * xcv;
        float z = xz[row * 2048 + DM + d];
        y *= z / (1.0f + __expf(-z));                            // * silu(z)
        yb[row * DM + d] = f32_to_bf16(y);
    }
}

// ---------------------------------------------------------------------------
// LayerNorm over D=1024, in place; also writes bf16 copy for next GEMM.
// ---------------------------------------------------------------------------
__global__ __launch_bounds__(256)
void layernorm_kernel(float* __restrict__ h, const float* __restrict__ g,
                      const float* __restrict__ bt, u16* __restrict__ hb)
{
    __shared__ float red[2][256];
    size_t row = blockIdx.x;
    float* x = h + row * DM;
    float vals[4], s = 0, ss = 0;
    #pragma unroll
    for (int i = 0; i < 4; i++) {
        float v = x[threadIdx.x + i * 256];
        vals[i] = v; s += v; ss += v * v;
    }
    red[0][threadIdx.x] = s; red[1][threadIdx.x] = ss;
    __syncthreads();
    for (int o = 128; o > 0; o >>= 1) {
        if ((int)threadIdx.x < o) {
            red[0][threadIdx.x] += red[0][threadIdx.x + o];
            red[1][threadIdx.x] += red[1][threadIdx.x + o];
        }
        __syncthreads();
    }
    float mean = red[0][0] * (1.0f / DM);
    float var  = red[1][0] * (1.0f / DM) - mean * mean;
    float inv  = rsqrtf(var + 1e-5f);
    #pragma unroll
    for (int i = 0; i < 4; i++) {
        int c = threadIdx.x + i * 256;
        float v = (vals[i] - mean) * inv * g[c] + bt[c];
        x[c] = v;
        hb[row * DM + c] = f32_to_bf16(v);
    }
}

// ---------------------------------------------------------------------------
// De-normalize: out[b,p,v] = dec[b*6+v, p] * std[b,v] + mean[b,v]
// ---------------------------------------------------------------------------
__global__ void denorm_kernel(const float* __restrict__ dec, const float* __restrict__ stats,
                              float* __restrict__ out)
{
    int i = blockIdx.x * 256 + threadIdx.x;
    if (i >= BATCH * PRED * 2) return;
    int v = i & 1;
    int p = (i >> 1) % PRED;
    int b = i / (PRED * 2);
    out[i] = dec[(size_t)(b * 6 + v) * PRED + p] * stats[b * 4 + 2 + v] + stats[b * 4 + v];
}

// ---------------------------------------------------------------------------
// GEMM dispatch
// ---------------------------------------------------------------------------
enum GemmMode { G_PLAIN, G_BIAS, G_BIAS_BOTH, G_ACCUM, G_BIAS_RELU_BF16, G_BIAS_ACCUM };

static void launch_gemm(hipStream_t st, GemmMode m, const u16* A, const u16* W,
                        const float* bias, float* Cf, u16* Cb, int M, int N, int K)
{
    int kchunk = (K % 128 == 0) ? 128 : 64;   // K in {768,1024} -> 128 ; 64 -> 64
    dim3 g((N + 63) / 64, M / 128), b(256);
    switch (m) {
    case G_PLAIN:         wmma_gemm_kernel<false,false,false,true ,false><<<g,b,0,st>>>(A,W,bias,Cf,Cb,M,N,K,kchunk); break;
    case G_BIAS:          wmma_gemm_kernel<true ,false,false,true ,false><<<g,b,0,st>>>(A,W,bias,Cf,Cb,M,N,K,kchunk); break;
    case G_BIAS_BOTH:     wmma_gemm_kernel<true ,false,false,true ,true ><<<g,b,0,st>>>(A,W,bias,Cf,Cb,M,N,K,kchunk); break;
    case G_ACCUM:         wmma_gemm_kernel<false,false,true ,true ,false><<<g,b,0,st>>>(A,W,bias,Cf,Cb,M,N,K,kchunk); break;
    case G_BIAS_RELU_BF16:wmma_gemm_kernel<true ,true ,false,false,true ><<<g,b,0,st>>>(A,W,bias,Cf,Cb,M,N,K,kchunk); break;
    case G_BIAS_ACCUM:    wmma_gemm_kernel<true ,false,true ,true ,false><<<g,b,0,st>>>(A,W,bias,Cf,Cb,M,N,K,kchunk); break;
    }
}

// ---------------------------------------------------------------------------
extern "C" void kernel_launch(void* const* d_in, const int* in_sizes, int n_in,
                              void* d_out, int out_size, void* d_ws, size_t ws_size,
                              hipStream_t stream)
{
    const float* x_enc   = (const float*)d_in[0];
    const float* x_mark  = (const float*)d_in[1];
    const float* emb_W   = (const float*)d_in[4];
    const float* emb_b   = (const float*)d_in[5];
    const float* in_W    = (const float*)d_in[6];
    const float* conv_w  = (const float*)d_in[7];
    const float* conv_b  = (const float*)d_in[8];
    const float* xproj_W = (const float*)d_in[9];
    const float* dt_W    = (const float*)d_in[10];
    const float* dt_b    = (const float*)d_in[11];
    const float* A_log   = (const float*)d_in[12];
    const float* D_par   = (const float*)d_in[13];
    const float* out_W   = (const float*)d_in[14];
    const float* ln1_g   = (const float*)d_in[15];
    const float* ln1_b   = (const float*)d_in[16];
    const float* f_w1    = (const float*)d_in[17];
    const float* f_b1    = (const float*)d_in[18];
    const float* f_w2    = (const float*)d_in[19];
    const float* f_b2    = (const float*)d_in[20];
    const float* ln2_g   = (const float*)d_in[21];
    const float* ln2_b   = (const float*)d_in[22];
    const float* nf_g    = (const float*)d_in[23];
    const float* nf_b    = (const float*)d_in[24];
    const float* proj_W  = (const float*)d_in[25];
    const float* proj_b  = (const float*)d_in[26];

    char* ws = (char*)d_ws;
    size_t off = 0;
    auto arena = [&](size_t bytes) -> char* {
        char* p = ws + off;
        off = (off + bytes + 255) & ~(size_t)255;
        return p;
    };

    float* stats = (float*)arena(BATCH * 4 * 4);
    u16*   tok   = (u16*)  arena((size_t)R_TOK * KPAD * 2);
    float* h     = (float*)arena((size_t)R_TOK * DM * 4);
    u16*   hb    = (u16*)  arena((size_t)R_TOK * DM * 2);
    float* xz    = (float*)arena((size_t)R_TOK * 2048 * 4);
    float* xcf   = (float*)arena((size_t)R_TOK * DM * 4);
    u16*   xcb   = (u16*)  arena((size_t)R_TOK * DM * 2);
    float* dbc   = (float*)arena((size_t)R_TOK * 192 * 4);
    u16*   dtb   = (u16*)  arena((size_t)R_TOK * DTR * 2);
    float* dtl   = (float*)arena((size_t)R_TOK * DM * 4);
    u16*   yb    = (u16*)  arena((size_t)R_TOK * DM * 2);
    u16*   ffb   = (u16*)  arena((size_t)R_TOK * DM * 2);
    float* dec   = (float*)arena((size_t)R_TOK * PRED * 4);

    u16* wEmb = (u16*)arena((size_t)DM * KPAD * 2);
    u16* wIn  = (u16*)arena((size_t)6 * 2048 * DM * 2);
    u16* wXp  = (u16*)arena((size_t)6 * 192 * DM * 2);
    u16* wDt  = (u16*)arena((size_t)6 * DM * DTR * 2);
    u16* wOut = (u16*)arena((size_t)6 * DM * DM * 2);
    u16* wF1  = (u16*)arena((size_t)3 * DM * DM * 2);
    u16* wF2  = (u16*)arena((size_t)3 * DM * DM * 2);
    u16* wPr  = (u16*)arena((size_t)PRED * DM * 2);
    if (off > ws_size) return;   // workspace too small; nothing safe to do

    auto cvt = [&](const float* s, u16* d, int n) {
        to_bf16_kernel<<<(n + 255) / 256, 256, 0, stream>>>(s, d, n);
    };
    // one-shot weight conversions (deterministic, re-done every call)
    embw_pad_kernel<<<(DM * KPAD + 255) / 256, 256, 0, stream>>>(emb_W, wEmb);
    cvt(in_W,    wIn,  6 * 2048 * DM);
    cvt(xproj_W, wXp,  6 * 192 * DM);
    cvt(dt_W,    wDt,  6 * DM * DTR);
    cvt(out_W,   wOut, 6 * DM * DM);
    cvt(f_w1,    wF1,  3 * DM * DM);
    cvt(f_w2,    wF2,  3 * DM * DM);
    cvt(proj_W,  wPr,  PRED * DM);

    // RevIN + token build, then embedding GEMM -> h (f32) + hb (bf16)
    revin_tokens_kernel<<<BATCH, 256, 0, stream>>>(x_enc, x_mark, tok, stats);
    launch_gemm(stream, G_BIAS_BOTH, tok, wEmb, emb_b, h, hb, R_TOK, DM, KPAD);

    for (int l = 0; l < 3; l++) {
        for (int dir = 0; dir < 2; dir++) {
            int li = l * 2 + dir;
            // in_proj: xz = hb @ in_W^T   [R, 2048]
            launch_gemm(stream, G_PLAIN, hb, wIn + (size_t)li * 2048 * DM, nullptr,
                        xz, nullptr, R_TOK, 2048, DM);
            // depthwise conv over T + SiLU
            conv_silu_kernel<<<R_TOK * DM / 256, 256, 0, stream>>>(
                xz, conv_w + (size_t)li * DM * 2, conv_b + (size_t)li * DM, xcf, xcb, dir);
            // x_proj: dbc = xc @ xproj_W^T   [R, 192]
            launch_gemm(stream, G_PLAIN, xcb, wXp + (size_t)li * 192 * DM, nullptr,
                        dbc, nullptr, R_TOK, 192, DM);
            // dt: dtl = dbc[:, :64] @ dt_W^T + dt_b
            dt_slice_kernel<<<R_TOK * DTR / 256, 256, 0, stream>>>(dbc, dtb);
            launch_gemm(stream, G_BIAS, dtb, wDt + (size_t)li * DM * DTR,
                        dt_b + (size_t)li * DM, dtl, nullptr, R_TOK, DM, DTR);
            // selective scan + gate -> yb (bf16)
            scan_kernel<<<dim3(DM / 256, BATCH), 256, 0, stream>>>(
                dbc, dtl, xcf, xz, A_log + (size_t)li * DM * DS,
                D_par + (size_t)li * DM, yb, dir);
            // out_proj accumulated into residual: h += yb @ out_W^T
            launch_gemm(stream, G_ACCUM, yb, wOut + (size_t)li * DM * DM, nullptr,
                        h, nullptr, R_TOK, DM, DM);
        }
        layernorm_kernel<<<R_TOK, 256, 0, stream>>>(h, ln1_g + l * DM, ln1_b + l * DM, hb);
        // FFN: relu(hb@w1^T+b1) -> ffb (bf16); h += ffb@w2^T + b2
        launch_gemm(stream, G_BIAS_RELU_BF16, hb, wF1 + (size_t)l * DM * DM,
                    f_b1 + l * DM, nullptr, ffb, R_TOK, DM, DM);
        launch_gemm(stream, G_BIAS_ACCUM, ffb, wF2 + (size_t)l * DM * DM,
                    f_b2 + l * DM, h, nullptr, R_TOK, DM, DM);
        layernorm_kernel<<<R_TOK, 256, 0, stream>>>(h, ln2_g + l * DM, ln2_b + l * DM, hb);
    }

    // final norm + projection + de-normalize
    layernorm_kernel<<<R_TOK, 256, 0, stream>>>(h, nf_g, nf_b, hb);
    launch_gemm(stream, G_BIAS, hb, wPr, proj_b, dec, nullptr, R_TOK, PRED, DM);
    denorm_kernel<<<(BATCH * PRED * 2 + 255) / 256, 256, 0, stream>>>(dec, stats, (float*)d_out);
}
